// PerceptNet_48679159333131
// MI455X (gfx1250) — matrix operand order repaired and checked by reading, not compiled, not run on hardware
//
#include <hip/hip_runtime.h>

// ---------------------------------------------------------------------------
// PerceptNet pipeline for MI455X (gfx1250, wave32, WMMA).
// Gabor 1->32 conv and 32x32 channel coupling run on v_wmma_f32_16x16x32_f16.
// Spatial Gaussian pooling uses exact separability. k3b input tiles are
// staged to LDS with the Tensor Data Mover (TENSORcnt); k4a windows use
// async LDS loads (ASYNCcnt).
// ---------------------------------------------------------------------------

#define USE_ASYNC_LDS 1
#define USE_TDM 1

typedef __attribute__((ext_vector_type(16))) _Float16 v16h;
typedef __attribute__((ext_vector_type(8)))  float    v8f;
typedef __attribute__((ext_vector_type(4)))  unsigned tdm_u4;
typedef __attribute__((ext_vector_type(8)))  int      tdm_i8;
typedef __attribute__((ext_vector_type(4)))  int      tdm_i4;

#define PI_F 3.14159265358979323846f
#define XM_F 0.328125f          // K/FS/2 = 21/32/2

struct U32x8 { uint4 lo, hi; };

__device__ __forceinline__ int refl(int t, int n) {
    if (t < 0) t = -1 - t;
    else if (t >= n) t = 2 * n - 1 - t;
    return t;
}
// A-matrix (16x32 f16) K-offset for (vgpr v, half hf); lane>=16 adds +8.
__device__ __forceinline__ int koffAlo(int v, int hf) {
    return (v < 4) ? (2 * v + hf) : (16 + 2 * (v - 4) + hf);
}
// B-matrix (32x16 f16) K-offset (same order as A; lane>=16 adds +8)
__device__ __forceinline__ int koffB(int v, int hf, bool hi) {
    return (hi ? 8 : 0) + ((v < 4) ? (2 * v + hf) : (16 + 2 * (v - 4) + hf));
}
__device__ __forceinline__ unsigned pack2h(float a, float b) {
    _Float16 ha = (_Float16)a, hb = (_Float16)b;
    unsigned short ua = __builtin_bit_cast(unsigned short, ha);
    unsigned short ub = __builtin_bit_cast(unsigned short, hb);
    return (unsigned)ua | ((unsigned)ub << 16);
}
__device__ __forceinline__ float gdn1f(float v, float g, float b) {
    return v * rsqrtf(b + g * v * v + 1e-6f);
}

// ---------------------------------------------------------------------------
// K0: generate module constants on device.
// ---------------------------------------------------------------------------
__global__ void k0_consts(const float* __restrict__ gabor_A,
                          float* __restrict__ gab_w,      // [32][441]
                          float* __restrict__ gs1,        // [21]
                          unsigned* __restrict__ packMix) // [2][32][8]
{
    int bi = blockIdx.x;
    int tid = threadIdx.x;
    if (bi < 32) {
        __shared__ float gv[441];
        __shared__ float red[256];
        int ch = bi, fi = ch >> 3, o = ch & 7;
        float f = (float)(2 << fi);
        float th = (float)o * PI_F / 8.f;
        float sig = 0.4f / f;
        float inv2s2 = 1.f / (2.f * sig * sig);
        float ct = cosf(th), st = sinf(th);
        for (int idx = tid; idx < 441; idx += 256) {
            int i = idx / 21, j = idx % 21;
            float xv = (float)i * (1.f / 32.f) - XM_F;
            float yv = (float)j * (1.f / 32.f) - XM_F;
            float xr = xv * ct + yv * st;
            gv[idx] = expf(-(xv * xv + yv * yv) * inv2s2) * cosf(2.f * PI_F * f * xr);
        }
        __syncthreads();
        float p = 0.f;
        for (int idx = tid; idx < 441; idx += 256) p += gv[idx];
        red[tid] = p; __syncthreads();
        for (int s = 128; s > 0; s >>= 1) {
            if (tid < s) red[tid] += red[tid + s];
            __syncthreads();
        }
        float mean = red[0] / 441.f;
        __syncthreads();
        for (int idx = tid; idx < 441; idx += 256) gv[idx] -= mean;
        __syncthreads();
        p = 0.f;
        for (int idx = tid; idx < 441; idx += 256) p += gv[idx] * gv[idx];
        red[tid] = p; __syncthreads();
        for (int s = 128; s > 0; s >>= 1) {
            if (tid < s) red[tid] += red[tid + s];
            __syncthreads();
        }
        float scale = rsqrtf(red[0] + 1e-12f) * gabor_A[ch];
        for (int idx = tid; idx < 441; idx += 256)
            gab_w[ch * 441 + idx] = gv[idx] * scale;
    } else {
        __shared__ float u[21];
        __shared__ float wc[1024];
        __shared__ float snorm;
        if (tid < 21) {
            float xv = (float)tid * (1.f / 32.f) - XM_F;
            u[tid] = expf(-xv * xv / 0.02f);   // sigma=0.1
        }
        __syncthreads();
        if (tid == 0) {
            float s = 0.f;
            for (int i = 0; i < 21; ++i) s += u[i];
            snorm = 1.f / s;
        }
        __syncthreads();
        if (tid < 21) gs1[tid] = u[tid] * snorm;
        for (int idx = tid; idx < 1024; idx += 256) {
            int c = idx >> 5, d = idx & 31;
            float fc = (float)(2 << (c >> 3)), fd = (float)(2 << (d >> 3));
            float tc = (float)(c & 7) * PI_F / 8.f, td = (float)(d & 7) * PI_F / 8.f;
            float df = fc - fd;
            float dt = fabsf(tc - td); dt = fminf(dt, PI_F - dt);
            float st2 = (PI_F / 4.f) * (PI_F / 4.f);
            wc[idx] = expf(-df * df / 32.f) * expf(-dt * dt / (2.f * st2));
        }
        __syncthreads();
        for (int idx = tid; idx < 512; idx += 256) {
            int ng = idx >> 8, rem = idx & 255, lane = rem >> 3, v = rem & 7;
            int n = lane & 15; bool hi = lane & 16;
            int c = ng * 16 + n;
            int k0 = koffB(v, 0, hi), k1 = koffB(v, 1, hi);
            packMix[idx] = pack2h(wc[c * 32 + k0], wc[c * 32 + k1]);
        }
    }
}

// K0b: pack Gabor weights into WMMA B fragments [2][14][32][8] u32
__global__ void k0b_packgab(const float* __restrict__ gab_w,
                            unsigned* __restrict__ packGab)
{
    int bi = blockIdx.x;
    int ng = bi / 14, kc = bi % 14;
    int lane = threadIdx.x;
    int n = lane & 15; bool hi = lane & 16;
    int ch = ng * 16 + n, kbase = kc * 32;
    for (int v = 0; v < 8; ++v) {
        int k0 = kbase + koffB(v, 0, hi);
        int k1 = kbase + koffB(v, 1, hi);
        float a = (k0 < 441) ? gab_w[ch * 441 + k0] : 0.f;
        float b = (k1 < 441) ? gab_w[ch * 441 + k1] : 0.f;
        packGab[((size_t)(ng * 14 + kc) * 32 + lane) * 8 + v] = pack2h(a, b);
    }
}

// ---------------------------------------------------------------------------
// K1: GDN1 -> 2x2 maxpool -> GDN2 into padded patch layout p_pad[256][84][84][3]
// ---------------------------------------------------------------------------
__global__ void k1_gdnpool(const float* __restrict__ in,
                           const float* __restrict__ g1, const float* __restrict__ b1,
                           const float* __restrict__ g2, const float* __restrict__ b2,
                           float* __restrict__ p_pad)
{
    int idx = blockIdx.x * 256 + threadIdx.x;          // 5,419,008 total (exact)
    int c = idx % 3;
    int x = (idx / 3) % 84;
    int y = (idx / 252) % 84;
    int pi = idx / 21168;
    int b = pi & 15, p2 = (pi >> 4) & 3, p1 = pi >> 6;
    int sy = refl(y - 10, 64), sx = refl(x - 10, 64);
    int hh = p1 * 64 + sy, ww = p2 * 64 + sx;
    const float* base = in + ((size_t)(b * 512 + 2 * hh) * 512 + 2 * ww) * 3 + c;
    float ga = g1[c], be = b1[c];
    float v0 = gdn1f(base[0], ga, be);
    float v1 = gdn1f(base[3], ga, be);
    float v2 = gdn1f(base[1536], ga, be);
    float v3 = gdn1f(base[1539], ga, be);
    float m = fmaxf(fmaxf(v0, v1), fmaxf(v2, v3));
    p_pad[idx] = gdn1f(m, g2[c], b2[c]);
}

// ---------------------------------------------------------------------------
// K2a: surround signal s = relu(p - gauss_SAME(p)), per (patch, channel).
// ---------------------------------------------------------------------------
__global__ void k2a_surround(const float* __restrict__ p_pad,
                             const float* __restrict__ inrf_g,
                             float* __restrict__ s_buf)
{
    __shared__ float tile[7056];  // 84x84
    __shared__ float gk[441];
    int pi = blockIdx.x / 3, c = blockIdx.x % 3;
    const float* src = p_pad + (size_t)pi * 21168 + c;
    for (int idx = threadIdx.x; idx < 7056; idx += 256) tile[idx] = src[(size_t)idx * 3];
    for (int idx = threadIdx.x; idx < 441; idx += 256) gk[idx] = inrf_g[idx];
    __syncthreads();
    for (int o = threadIdx.x; o < 7056; o += 256) {
        int y = o / 84, x = o % 84;
        float acc = 0.f;
        for (int i = 0; i < 21; ++i) {
            int yy = y + i - 10;
            if ((unsigned)yy < 84u) {
                const float* tr = tile + yy * 84;
                const float* gr = gk + i * 21;
                for (int j = 0; j < 21; ++j) {
                    int xx = x + j - 10;
                    if ((unsigned)xx < 84u) acc += gr[j] * tr[xx];
                }
            }
        }
        s_buf[(size_t)pi * 21168 + (size_t)o * 3 + c] = fmaxf(tile[o] - acc, 0.f);
    }
}

// ---------------------------------------------------------------------------
// K2b: INRF VALID convs: y = lam * (w*p) - (m*s), 64x64 per patch, written
// into the interior of y_pad[16][276][276]. Tiles held as f16 in LDS (<64KB).
// ---------------------------------------------------------------------------
__global__ void __launch_bounds__(256) k2b_inrf(
    const float* __restrict__ p_pad, const float* __restrict__ s_buf,
    const float* __restrict__ inrf_w, const float* __restrict__ inrf_m,
    const float* __restrict__ lam_p, float* __restrict__ y_pad)
{
    __shared__ _Float16 pt[13104];   // 84 x 52 x 3
    __shared__ _Float16 st2[13104];
    __shared__ float wk[1323];
    __shared__ float mk[1323];
    int pi = blockIdx.x >> 1, half = blockIdx.x & 1;
    int ox0 = half * 32;
    const float* pb = p_pad + (size_t)pi * 21168 + ox0 * 3;
    const float* sb = s_buf + (size_t)pi * 21168 + ox0 * 3;
    for (int li = threadIdx.x; li < 13104; li += 256) {
        int gi = li + (li / 156) * 96;           // (y*84+xx)*3+c with xx<52
        pt[li]  = (_Float16)pb[gi];
        st2[li] = (_Float16)sb[gi];
    }
    for (int li = threadIdx.x; li < 1323; li += 256) {
        wk[li] = inrf_w[li];
        mk[li] = inrf_m[li];
    }
    __syncthreads();
    float lam = lam_p[0];
    int ox = threadIdx.x & 31;
    int oy0 = (threadIdx.x >> 5) * 8;
    float accw[8], accm[8];
#pragma unroll
    for (int k = 0; k < 8; ++k) { accw[k] = 0.f; accm[k] = 0.f; }
    for (int i = 0; i < 21; ++i)
        for (int j = 0; j < 21; ++j)
            for (int c = 0; c < 3; ++c) {
                float wv = wk[(i * 21 + j) * 3 + c];
                float mv = mk[(i * 21 + j) * 3 + c];
                int t0 = ((oy0 + i) * 52 + ox + j) * 3 + c;
#pragma unroll
                for (int k = 0; k < 8; ++k) {
                    int t = t0 + k * 156;
                    accw[k] += wv * (float)pt[t];
                    accm[k] += mv * (float)st2[t];
                }
            }
    int b = pi & 15, p2 = (pi >> 4) & 3, p1 = pi >> 6;
#pragma unroll
    for (int k = 0; k < 8; ++k) {
        int gh = p1 * 64 + oy0 + k;
        int gw = p2 * 64 + ox0 + ox;
        y_pad[((size_t)b * 276 + (10 + gh)) * 276 + (10 + gw)] = lam * accw[k] - accm[k];
    }
}

// K3a: fill symmetric-reflect border of y_pad from its interior.
__global__ void k3a_pad(float* __restrict__ y_pad)
{
    int idx = blockIdx.x * 256 + threadIdx.x;  // 1,218,816 (exact)
    int w = idx % 276, h = (idx / 276) % 276, b = idx / 76176;
    if (h >= 10 && h < 266 && w >= 10 && w < 266) return;
    int sh = refl(h - 10, 256), sw = refl(w - 10, 256);
    y_pad[idx] = y_pad[((size_t)b * 276 + 10 + sh) * 276 + 10 + sw];
}

// ---------------------------------------------------------------------------
// K3b: Gabor conv as im2col GEMM via WMMA.
// Block = 128 thr (4 waves) = 64 pixels x 32 channels. Wave w owns M-tile w
// (16 pixels) and runs BOTH channel groups, so each A fragment feeds 2 WMMAs.
// kc loop fully unrolled: every im2col (i,j) offset is a compile-time
// constant. Input tile (21x84 f32, row stride 276) staged to LDS by the
// Tensor Data Mover (one instruction, TENSORcnt-tracked).
// ---------------------------------------------------------------------------
__global__ void __launch_bounds__(128) k3b_gabor(
    const float* __restrict__ y_pad, const unsigned* __restrict__ packGab,
    float* __restrict__ out)
{
    __shared__ float ys[1764];   // 21 rows x 84 cols
    int bi = blockIdx.x;
    int wblk = bi & 3, h = (bi >> 2) & 255, b = bi >> 10;
    int w0 = wblk * 64;
    const float* src = y_pad + ((size_t)b * 276 + h) * 276 + w0;
#if USE_TDM
    if (threadIdx.x < 32) {
        unsigned lds0 = (unsigned)(uintptr_t)(&ys[0]);
        unsigned long long ga = (unsigned long long)(uintptr_t)src;
        // D# group 0: count=1, lds_addr, global_addr[56:0], type=2
        tdm_u4 g0;
        g0.x = 1u;
        g0.y = lds0;
        g0.z = (unsigned)(ga & 0xFFFFFFFFu);
        g0.w = (unsigned)((ga >> 32) & 0x01FFFFFFu) | (2u << 30);
        // D# group 1: mask=0, data_size=2 (4B), tensor_dim0=276,
        // tensor_dim1=4096, tile_dim0=84, tile_dim1=21, tile_dim2=0,
        // tensor_dim0_stride=276, tensor_dim1_stride=0.
        tdm_i8 g1;
        g1[0] = 0x00020000;
        g1[1] = (int)(276u << 16);          // tensor_dim0[15:0] << 16
        g1[2] = (int)(4096u << 16);         // td0 hi=0 | tensor_dim1[15:0]<<16
        g1[3] = (int)(84u << 16);           // td1 hi=0 | tile_dim0
        g1[4] = 21;                         // tile_dim1 | tile_dim2=0
        g1[5] = 276;                        // tensor_dim0_stride[31:0]
        g1[6] = 0;
        g1[7] = 0;
        tdm_i4 gz = {0, 0, 0, 0};
#if defined(__clang_major__) && (__clang_major__ >= 23)
        tdm_i8 gz8 = {0, 0, 0, 0, 0, 0, 0, 0};
        __builtin_amdgcn_tensor_load_to_lds(g0, g1, gz, gz, gz8, 0);
#else
        __builtin_amdgcn_tensor_load_to_lds(g0, g1, gz, gz, 0);
#endif
        __builtin_amdgcn_s_wait_tensorcnt(0);
    }
#else
    unsigned lds0 = (unsigned)(uintptr_t)(&ys[0]);
    for (int li = threadIdx.x; li < 1764; li += 128) {
        int r = li / 84, cx = li - r * 84;
        const float* g = src + (size_t)r * 276 + cx;
        unsigned loff = lds0 + 4u * (unsigned)li;
        asm volatile("global_load_async_to_lds_b32 %0, %1, off"
                     :: "v"(loff), "v"(g) : "memory");
    }
    asm volatile("s_wait_asynccnt 0x0" ::: "memory");
#endif
    __syncthreads();

    int lane = threadIdx.x & 31;
    int mt = threadIdx.x >> 5;          // M-tile 0..3
    int m = lane & 15;
    bool hi = lane & 16;
    int mcol = mt * 16 + m;
    v8f acc0 = {0.f, 0.f, 0.f, 0.f, 0.f, 0.f, 0.f, 0.f};
    v8f acc1 = {0.f, 0.f, 0.f, 0.f, 0.f, 0.f, 0.f, 0.f};
#pragma unroll
    for (int kc = 0; kc < 14; ++kc) {
        v16h a;
#pragma unroll
        for (int v = 0; v < 8; ++v) {
#pragma unroll
            for (int hf = 0; hf < 2; ++hf) {
                const int klo = kc * 32 + koffAlo(v, hf);  // always < 441
                const int khi = klo + 8;
                const int offlo = (klo / 21) * 84 + (klo % 21);
                float val;
                if (khi < 441) {
                    const int offhi = (khi / 21) * 84 + (khi % 21);
                    val = ys[(hi ? offhi : offlo) + mcol];
                } else {
                    val = hi ? 0.f : ys[offlo + mcol];
                }
                a[2 * v + hf] = (_Float16)val;
            }
        }
        {
            const uint4* bp = reinterpret_cast<const uint4*>(
                packGab + ((size_t)(kc * 32 + lane)) * 8);           // ng=0
            U32x8 u; u.lo = bp[0]; u.hi = bp[1];
            v16h bf = __builtin_bit_cast(v16h, u);
            acc0 = __builtin_amdgcn_wmma_f32_16x16x32_f16(
                false, a, false, bf, (short)0, acc0, false, false);
        }
        {
            const uint4* bp = reinterpret_cast<const uint4*>(
                packGab + ((size_t)((14 + kc) * 32 + lane)) * 8);    // ng=1
            U32x8 u; u.lo = bp[0]; u.hi = bp[1];
            v16h bf = __builtin_bit_cast(v16h, u);
            acc1 = __builtin_amdgcn_wmma_f32_16x16x32_f16(
                false, a, false, bf, (short)0, acc1, false, false);
        }
    }
    int moff = hi ? 8 : 0;
    size_t rowbase = ((size_t)b * 256 + h) * 256;
#pragma unroll
    for (int r = 0; r < 8; ++r) {
        int w = w0 + mt * 16 + r + moff;
        out[(rowbase + w) * 32 + m] = acc0[r];
        out[(rowbase + w) * 32 + 16 + m] = acc1[r];
    }
}

// ---------------------------------------------------------------------------
// K4a: horizontal pass of separable Gaussian (sigma=0.1) on energy gab^2,
// symmetric reflection. Block = 64 pixels x 32 channels; the 84x32 window is
// staged to LDS with async LDS loads, squared in place, then convolved.
// ---------------------------------------------------------------------------
__global__ void __launch_bounds__(256) k4a_hblur(
    const float* __restrict__ gab, const float* __restrict__ gs1,
    _Float16* __restrict__ t16)
{
    __shared__ float es[2688];   // 84 pixels x 32 channels
    __shared__ float gs[21];
    if (threadIdx.x < 21) gs[threadIdx.x] = gs1[threadIdx.x];
    int bi = blockIdx.x;
    int wblk = bi & 3, h = (bi >> 2) & 255, b = bi >> 10;
    int w0 = wblk * 64;
    size_t rowbase = (((size_t)b * 256 + h) << 13);   // *256*32
#if USE_ASYNC_LDS
    unsigned lds0 = (unsigned)(uintptr_t)(&es[0]);
    for (int li = threadIdx.x; li < 2688; li += 256) {
        int p = li >> 5, c = li & 31;
        int wr = refl(w0 + p - 10, 256);
        const float* g = gab + rowbase + ((size_t)wr << 5) + c;
        unsigned loff = lds0 + 4u * (unsigned)li;
        asm volatile("global_load_async_to_lds_b32 %0, %1, off"
                     :: "v"(loff), "v"(g) : "memory");
    }
    asm volatile("s_wait_asynccnt 0x0" ::: "memory");
#else
    for (int li = threadIdx.x; li < 2688; li += 256) {
        int p = li >> 5, c = li & 31;
        int wr = refl(w0 + p - 10, 256);
        es[li] = gab[rowbase + ((size_t)wr << 5) + c];
    }
#endif
    __syncthreads();
    for (int li = threadIdx.x; li < 2688; li += 256) {
        float v = es[li];
        es[li] = v * v;
    }
    __syncthreads();
#pragma unroll
    for (int k = 0; k < 8; ++k) {
        int vi = threadIdx.x + 256 * k;          // 2048 outputs
        int wl = vi >> 5, c = vi & 31;
        float acc = 0.f;
#pragma unroll
        for (int j = 0; j < 21; ++j)
            acc += gs[j] * es[((wl + j) << 5) + c];
        t16[rowbase + ((size_t)(w0 + wl) << 5) + c] = (_Float16)acc;
    }
}

// ---------------------------------------------------------------------------
// K4b: vertical Gaussian pass -> sp[64][32] in LDS, then 32x32 channel
// coupling via WMMA (K=32), then final GDN. Updates gab (= d_out) in place;
// each block only touches its own pixels.
// ---------------------------------------------------------------------------
__global__ void __launch_bounds__(128) k4b_final(
    const _Float16* __restrict__ t16, const float* __restrict__ gs1,
    const unsigned* __restrict__ packMix,
    const float* __restrict__ gdn_g, const float* __restrict__ gdn_b,
    float* __restrict__ out)
{
    __shared__ float sp[64 * 33];
    __shared__ float gs[21];
    if (threadIdx.x < 21) gs[threadIdx.x] = gs1[threadIdx.x];
    __syncthreads();
    int bi = blockIdx.x;
    int wblk = bi & 3, h = (bi >> 2) & 255, b = bi >> 10;
    int w0 = wblk * 64;
    for (int k = 0; k < 16; ++k) {
        int vi = threadIdx.x + 128 * k;
        int wl = vi >> 5, c = vi & 31;
        float acc = 0.f;
        for (int i = 0; i < 21; ++i) {
            int hr = refl(h + i - 10, 256);
            acc += gs[i] * (float)t16[(((size_t)b * 256 + hr) * 256 + (w0 + wl)) * 32 + c];
        }
        sp[wl * 33 + c] = acc;
    }
    __syncthreads();
    int lane = threadIdx.x & 31;
    int mtile = threadIdx.x >> 5;
    int n = lane & 15;
    bool hi = lane & 16;
    int abase = (mtile * 16 + n) * 33 + (hi ? 8 : 0);
    v16h a;
#pragma unroll
    for (int v = 0; v < 8; ++v) {
#pragma unroll
        for (int hf = 0; hf < 2; ++hf) {
            const int base = koffAlo(v, hf);     // compile-time constant
            a[2 * v + hf] = (_Float16)sp[abase + base];
        }
    }
#pragma unroll
    for (int ng = 0; ng < 2; ++ng) {
        const uint4* bp = reinterpret_cast<const uint4*>(packMix + ((size_t)(ng * 32 + lane)) * 8);
        U32x8 u; u.lo = bp[0]; u.hi = bp[1];
        v16h bf = __builtin_bit_cast(v16h, u);
        v8f acc = {0.f, 0.f, 0.f, 0.f, 0.f, 0.f, 0.f, 0.f};
        acc = __builtin_amdgcn_wmma_f32_16x16x32_f16(
            false, a, false, bf, (short)0, acc, false, false);
        int ch = ng * 16 + n;
        float gg = gdn_g[ch], bb = gdn_b[ch];
        int moff = hi ? 8 : 0;
#pragma unroll
        for (int r = 0; r < 8; ++r) {
            int w = w0 + mtile * 16 + r + moff;
            size_t oi = (((size_t)b * 256 + h) * 256 + w) * 32 + ch;
            float g = out[oi];
            out[oi] = g * rsqrtf(bb + gg * acc[r] + 1e-6f);
        }
    }
}

// ---------------------------------------------------------------------------
// Host launcher.  Workspace layout (bytes), total ~110 MiB:
//   t16[16,256,256,32]f16 : 67,108,864   p_pad[256,84,84,3]f32 : 21,676,032
//   s_buf                 : 21,676,032   y_pad[16,276,276]f32  :  4,875,264
//   gab_w[32,441]f32      :     56,448   packGab[2,14,32,8]u32 :     28,672
//   packMix[2,32,8]u32    :      2,048   gs1[21]f32            :        256
// ---------------------------------------------------------------------------
extern "C" void kernel_launch(void* const* d_in, const int* in_sizes, int n_in,
                              void* d_out, int out_size, void* d_ws, size_t ws_size,
                              hipStream_t stream)
{
    const float* in       = (const float*)d_in[0];
    const float* gamma1   = (const float*)d_in[1];
    const float* beta1    = (const float*)d_in[2];
    const float* gamma2   = (const float*)d_in[3];
    const float* beta2    = (const float*)d_in[4];
    const float* inrf_w   = (const float*)d_in[5];
    const float* inrf_m   = (const float*)d_in[6];
    const float* inrf_g   = (const float*)d_in[7];
    const float* inrf_lam = (const float*)d_in[8];
    const float* gabor_A  = (const float*)d_in[9];
    const float* gdn_g    = (const float*)d_in[10];
    const float* gdn_b    = (const float*)d_in[11];
    float* out = (float*)d_out;

    char* ws = (char*)d_ws;
    size_t off = 0;
    _Float16* t16    = (_Float16*)(ws + off); off += 67108864;
    float*    p_pad  = (float*)   (ws + off); off += 21676032;
    float*    s_buf  = (float*)   (ws + off); off += 21676032;
    float*    y_pad  = (float*)   (ws + off); off += 4875264;
    float*    gab_w  = (float*)   (ws + off); off += 56448;
    unsigned* packGab= (unsigned*)(ws + off); off += 28672;
    unsigned* packMix= (unsigned*)(ws + off); off += 2048;
    float*    gs1    = (float*)   (ws + off); off += 256;

    k0_consts   <<<33,    256, 0, stream>>>(gabor_A, gab_w, gs1, packMix);
    k0b_packgab <<<28,     32, 0, stream>>>(gab_w, packGab);
    k1_gdnpool  <<<21168, 256, 0, stream>>>(in, gamma1, beta1, gamma2, beta2, p_pad);
    k2a_surround<<<768,   256, 0, stream>>>(p_pad, inrf_g, s_buf);
    k2b_inrf    <<<512,   256, 0, stream>>>(p_pad, s_buf, inrf_w, inrf_m, inrf_lam, y_pad);
    k3a_pad     <<<4761,  256, 0, stream>>>(y_pad);
    k3b_gabor   <<<16384, 128, 0, stream>>>(y_pad, packGab, out);
    k4a_hblur   <<<16384, 256, 0, stream>>>(out, gs1, t16);
    k4b_final   <<<16384, 128, 0, stream>>>(t16, gs1, packMix, gdn_g, gdn_b, out);
}
